// RQVAE_3513283248284
// MI455X (gfx1250) — matrix operand order repaired and checked by reading, not compiled, not run on hardware
//
#include <hip/hip_runtime.h>

typedef __attribute__((ext_vector_type(16))) _Float16 v16h;
typedef __attribute__((ext_vector_type(8)))  _Float16 v8h;
typedef __attribute__((ext_vector_type(4)))  _Float16 v4h;
typedef __attribute__((ext_vector_type(8)))  float    v8f;
typedef __attribute__((ext_vector_type(4)))  int      v4i;

#define NCB     4
#define KCODES  256
#define DIM     384
#define BATCHSZ 131072
#define MT      64            // rows per block tile
#define RES_STR 392           // padded LDS stride (halves) for residual tile
#define CB_STR  40            // padded LDS stride (halves) for codebook chunk
#define KCHUNKS 12            // 384 / 32
#define CHUNKH  (KCODES * CB_STR)   // halves per chunk buffer

static constexpr int SMEM_BYTES =
    MT * RES_STR * 2 +        // residual f16 tile       (50176 B)
    2 * CHUNKH * 2 +          // double-buffered chunk   (40960 B)
    MT * 2 * 4 +              // redval                  (  512 B)
    MT * 2 * 4 +              // redidx                  (  512 B)
    MT * 4;                   // rowcode                 (  256 B)  total 92416 B

// -------------------------------------------------- async global->LDS support

#if defined(__has_builtin)
#if __has_builtin(__builtin_amdgcn_global_load_async_to_lds_b128)
#define HAVE_ASYNC_LDS 1
#endif
#endif
#ifndef HAVE_ASYNC_LDS
#define HAVE_ASYNC_LDS 0
#endif

typedef __attribute__((address_space(1))) v4i gv4i;   // global int4
typedef __attribute__((address_space(3))) v4i lv4i;   // LDS int4

__device__ __forceinline__ void async_ld_b128(const _Float16* g, _Float16* l) {
#if HAVE_ASYNC_LDS
    __builtin_amdgcn_global_load_async_to_lds_b128(
        (gv4i*)(unsigned long long)g,
        (lv4i*)(unsigned)(unsigned long long)l, 0, 0);
#else
    (void)g; (void)l;
#endif
}

__device__ __forceinline__ void wait_async(int n) {
#if HAVE_ASYNC_LDS
#if __has_builtin(__builtin_amdgcn_s_wait_asynccnt)
    if (n == 0) __builtin_amdgcn_s_wait_asynccnt(0);
    else        __builtin_amdgcn_s_wait_asynccnt(4);
#else
    if (n == 0) asm volatile("s_wait_asynccnt 0x0" ::: "memory");
    else        asm volatile("s_wait_asynccnt 0x4" ::: "memory");
#endif
#else
    (void)n;
#endif
}

// copy one 256-code x 32-half codebook chunk into an LDS buffer (thread==code)
__device__ __forceinline__ void issue_chunk(const _Float16* __restrict__ cbg,
                                            _Float16* __restrict__ dstbuf,
                                            int tid, int kk) {
    const _Float16* g = cbg + tid * DIM + kk;
    _Float16*       d = dstbuf + tid * CB_STR;
#if HAVE_ASYNC_LDS
    async_ld_b128(g,      d);
    async_ld_b128(g + 8,  d + 8);
    async_ld_b128(g + 16, d + 16);
    async_ld_b128(g + 24, d + 24);
#else
    const uint4* s4 = (const uint4*)g;
    uint4*       d4 = (uint4*)d;
    d4[0] = s4[0]; d4[1] = s4[1]; d4[2] = s4[2]; d4[3] = s4[3];
#endif
}

// ---------------------------------------------------------------- prep kernels

__global__ void k_convert_cb(const float* __restrict__ src, _Float16* __restrict__ dst) {
    int i = blockIdx.x * 256 + threadIdx.x;          // over (NCB*KCODES*DIM)/4
    const float4* s4 = (const float4*)src;
    float4 v = s4[i];
    v4h h;
    h[0] = (_Float16)v.x; h[1] = (_Float16)v.y;
    h[2] = (_Float16)v.z; h[3] = (_Float16)v.w;
    *(v4h*)(dst + (long)i * 4) = h;
}

__global__ void k_halfnorm(const float* __restrict__ cb, float* __restrict__ hn) {
    int idx = blockIdx.x * 256 + threadIdx.x;        // 0..NCB*KCODES-1
    const float* r = cb + (long)idx * DIM;
    float s = 0.f;
    for (int j = 0; j < DIM; ++j) s += r[j] * r[j];
    hn[idx] = 0.5f * s;
}

// ---------------------------------------------------------------- main stage

__global__ void __launch_bounds__(256)
k_rq_step(const float* __restrict__ x, const float* __restrict__ cbf32,
          const _Float16* __restrict__ cbf16, const float* __restrict__ halfnorm,
          float* __restrict__ recon, int* __restrict__ codesOut,
          float* __restrict__ lossAccum, int cbi)
{
    extern __shared__ char smem[];
    _Float16* resf16  = (_Float16*)smem;
    _Float16* cbchunk = resf16 + MT * RES_STR;        // 2 buffers of CHUNKH halves
    float*    redval  = (float*)(cbchunk + 2 * CHUNKH);
    int*      redidx  = (int*)(redval + MT * 2);
    int*      rowcode = redidx + MT * 2;

    const int  tid     = threadIdx.x;
    const long rowBase = (long)blockIdx.x * MT;

    // ---- A) residual = x - recon  ->  f16 tile in LDS
    const float4* x4 = (const float4*)x;
    float4*       r4 = (float4*)recon;
    for (int e = tid; e < MT * (DIM / 4); e += 256) {
        int  row = e / (DIM / 4);
        int  c4  = e % (DIM / 4);
        long g   = (rowBase + row) * (DIM / 4) + c4;
        float4 xv = x4[g];
        float4 rv = r4[g];
        v4h h;
        h[0] = (_Float16)(xv.x - rv.x); h[1] = (_Float16)(xv.y - rv.y);
        h[2] = (_Float16)(xv.z - rv.z); h[3] = (_Float16)(xv.w - rv.w);
        *(v4h*)(resf16 + row * RES_STR + c4 * 4) = h;
    }

    // ---- B) score = r . c  via WMMA f16 (f32 accum), async double-buffered K
    const int  lane    = tid & 31;
    const int  w       = tid >> 5;
    const int  m0      = (w & 3) * 16;    // 4 M-tiles of 16 rows
    const int  n0      = (w >> 2) * 128;  // 2 N-halves of 128 codes
    const int  l       = lane & 15;
    const bool hihalf  = lane >= 16;

    v8f zero = {0.f, 0.f, 0.f, 0.f, 0.f, 0.f, 0.f, 0.f};
    v8f acc[8];
#pragma unroll
    for (int nt = 0; nt < 8; ++nt) acc[nt] = zero;

    const _Float16* aBase = resf16 + (m0 + l) * RES_STR + (hihalf ? 8 : 0);
    const _Float16* cbg   = cbf16 + (long)cbi * KCODES * DIM;

    issue_chunk(cbg, cbchunk, tid, 0);                 // prefetch chunk 0 -> buf 0

    for (int kc = 0; kc < KCHUNKS; ++kc) {
        _Float16* cur = cbchunk + (kc & 1) * CHUNKH;
        if (kc + 1 < KCHUNKS) {
            issue_chunk(cbg, cbchunk + ((kc + 1) & 1) * CHUNKH, tid, (kc + 1) * 32);
            wait_async(4);                             // chunk kc landed, kc+1 in flight
        } else {
            wait_async(0);
        }
        __syncthreads();                               // chunk kc visible to all waves

        const int kk = kc * 32;
        v8h alo = *(const v8h*)(aBase + kk);
        v8h ahi = *(const v8h*)(aBase + kk + 16);
        v16h a  = __builtin_shufflevector(alo, ahi,
                    0,1,2,3,4,5,6,7,8,9,10,11,12,13,14,15);

        v16h bf[8];
#pragma unroll
        for (int nt = 0; nt < 8; ++nt) {
            const _Float16* bb = cur + (n0 + nt * 16 + l) * CB_STR + (hihalf ? 8 : 0);
            v8h blo = *(const v8h*)bb;
            v8h bhi = *(const v8h*)(bb + 16);
            bf[nt]  = __builtin_shufflevector(blo, bhi,
                        0,1,2,3,4,5,6,7,8,9,10,11,12,13,14,15);
        }
#pragma unroll
        for (int nt = 0; nt < 8; ++nt) {
            acc[nt] = __builtin_amdgcn_wmma_f32_16x16x32_f16(
                          false, a, false, bf[nt], (short)0, acc[nt], false, false);
        }
        __syncthreads();                               // done reading cur buffer
    }

    // ---- C) argmax over (score - ||c||^2/2)
    const float* hn = halfnorm + cbi * KCODES;
    float bv[8]; int bi[8];
#pragma unroll
    for (int v = 0; v < 8; ++v) { bv[v] = -3.4e38f; bi[v] = 0; }
#pragma unroll
    for (int nt = 0; nt < 8; ++nt) {
        int   code = n0 + nt * 16 + l;
        float h    = hn[code];
#pragma unroll
        for (int v = 0; v < 8; ++v) {
            float s = acc[nt][v] - h;
            if (s > bv[v] || (s == bv[v] && code < bi[v])) { bv[v] = s; bi[v] = code; }
        }
    }
#pragma unroll
    for (int v = 0; v < 8; ++v) {
#pragma unroll
        for (int m = 1; m < 16; m <<= 1) {
            float ov = __shfl_xor(bv[v], m, 16);
            int   oi = __shfl_xor(bi[v], m, 16);
            if (ov > bv[v] || (ov == bv[v] && oi < bi[v])) { bv[v] = ov; bi[v] = oi; }
        }
    }
    if (l == 0) {
        int half = n0 >> 7;
#pragma unroll
        for (int v = 0; v < 8; ++v) {
            int row = m0 + v + (hihalf ? 8 : 0);
            redval[row * 2 + half] = bv[v];
            redidx[row * 2 + half] = bi[v];
        }
    }
    __syncthreads();
    if (tid < MT) {
        float v0 = redval[tid * 2], v1 = redval[tid * 2 + 1];
        int   i0 = redidx[tid * 2], i1 = redidx[tid * 2 + 1];
        int code = (v1 > v0 || (v1 == v0 && i1 < i0)) ? i1 : i0;
        rowcode[tid] = code;
        codesOut[(rowBase + tid) * NCB + cbi] = code;
    }
    __syncthreads();

    // ---- D) gather chosen code, update recon, accumulate loss ssq
    const float4* cb4 = (const float4*)(cbf32 + (long)cbi * KCODES * DIM);
    float ssq = 0.f;
    for (int e = tid; e < MT * (DIM / 4); e += 256) {
        int  row  = e / (DIM / 4);
        int  c4   = e % (DIM / 4);
        long g    = (rowBase + row) * (DIM / 4) + c4;
        int  code = rowcode[row];
        float4 xv = x4[g];
        float4 rv = r4[g];
        float4 qv = cb4[(long)code * (DIM / 4) + c4];
        float nx = xv.x - rv.x - qv.x;
        float ny = xv.y - rv.y - qv.y;
        float nz = xv.z - rv.z - qv.z;
        float nw = xv.w - rv.w - qv.w;
        float4 o;
        o.x = xv.x - nx; o.y = xv.y - ny; o.z = xv.z - nz; o.w = xv.w - nw;
        r4[g] = o;
        ssq += nx * nx + ny * ny + nz * nz + nw * nw;
    }
#pragma unroll
    for (int m = 1; m < 32; m <<= 1) ssq += __shfl_xor(ssq, m, 32);
    if (lane == 0) redval[w] = ssq;
    __syncthreads();
    if (tid == 0) {
        float s = 0.f;
        for (int i = 0; i < 8; ++i) s += redval[i];
        atomicAdd(lossAccum, s);
    }
}

// ---------------------------------------------------------------- finalize

__global__ void k_finalize(const float* __restrict__ lossAccum, float* __restrict__ out) {
    // total_loss = sum_i 2 * mean((q_i - r_i)^2)  (stop_gradient is identity fwd)
    out[(size_t)BATCHSZ * DIM + (size_t)BATCHSZ * NCB] =
        lossAccum[0] * (2.0f / ((float)BATCHSZ * (float)DIM));
}

// ---------------------------------------------------------------- launch

extern "C" void kernel_launch(void* const* d_in, const int* in_sizes, int n_in,
                              void* d_out, int out_size, void* d_ws, size_t ws_size,
                              hipStream_t stream) {
    (void)in_sizes; (void)n_in; (void)ws_size;
    const float* x  = (const float*)d_in[0];
    const float* cb = (const float*)d_in[1];

    float* outf     = (float*)d_out;                         // [B*D] recon
    int*   codesOut = (int*)(outf + (size_t)BATCHSZ * DIM);  // [B*4] int32 codes
                                                             // + 1 float loss at end
    char*     ws        = (char*)d_ws;
    float*    lossAccum = (float*)ws;                                  // 256 B slot
    _Float16* cbf16     = (_Float16*)(ws + 256);                       // 768 KB
    float*    halfnorm  = (float*)(ws + 256 + (size_t)NCB * KCODES * DIM * 2);

    (void)hipMemsetAsync(d_out, 0, (size_t)out_size * sizeof(float), stream); // recon=0
    (void)hipMemsetAsync(d_ws, 0, 256, stream);                               // ssq=0

    k_convert_cb<<<(NCB * KCODES * DIM) / 4 / 256, 256, 0, stream>>>(cb, cbf16);
    k_halfnorm<<<(NCB * KCODES) / 256, 256, 0, stream>>>(cb, halfnorm);

    for (int cbi = 0; cbi < NCB; ++cbi) {
        k_rq_step<<<BATCHSZ / MT, 256, SMEM_BYTES, stream>>>(
            x, cb, cbf16, halfnorm, outf, codesOut, lossAccum, cbi);
    }
    k_finalize<<<1, 1, 0, stream>>>(lossAccum, outf);
}
